// ToMEBlock_83571473645970
// MI455X (gfx1250) — matrix operand order repaired and checked by reading, not compiled, not run on hardware
//
#include <hip/hip_runtime.h>
#include <hip/hip_bf16.h>

typedef unsigned short u16;
typedef unsigned int   u32;
typedef __attribute__((ext_vector_type(16))) __bf16 v16bf;
typedef __attribute__((ext_vector_type(8)))  __bf16 v8bf;
typedef __attribute__((ext_vector_type(8)))  float  v8f;

#define B_   32
#define T_   4096
#define C_   1024
#define D_   64
#define TA_  2048   // even tokens (sources)
#define TB_  2048   // odd tokens (destinations)
#define R_   1024   // merged count
#define TOUT_ (TA_ - R_ + TB_)   // 3072 output rows per batch
#define CHUNK_ 128               // B-panel tokens staged per LDS chunk

// ---------------------------------------------------------------------------
// K1: normalize metric rows (L2) and write bf16 panels A (even toks), B (odd)
// ---------------------------------------------------------------------------
__global__ __launch_bounds__(256) void tome_norm(const float* __restrict__ metric,
                                                 u16* __restrict__ Aws,
                                                 u16* __restrict__ Bws) {
    int tok = blockIdx.x * 256 + threadIdx.x;          // 0 .. B*T-1
    int b = tok >> 12;                                 // /4096
    int t = tok & 4095;
    const float4* row = (const float4*)(metric + (size_t)tok * D_);
    float vals[D_];
    float ss = 0.f;
#pragma unroll
    for (int i = 0; i < 16; ++i) {
        float4 f = row[i];
        vals[4*i+0] = f.x; vals[4*i+1] = f.y; vals[4*i+2] = f.z; vals[4*i+3] = f.w;
        ss += f.x*f.x + f.y*f.y + f.z*f.z + f.w*f.w;
    }
    float inv = rsqrtf(ss);
    u16* drow = ((t & 1) ? Bws : Aws) + ((size_t)b * TA_ + (t >> 1)) * D_;
#pragma unroll
    for (int i = 0; i < D_; i += 2) {   // pack 2 bf16 (RNE) per dword store
        u32 u0 = __float_as_uint(vals[i]   * inv);
        u32 u1 = __float_as_uint(vals[i+1] * inv);
        u0 = (u0 + 0x7FFFu + ((u0 >> 16) & 1u)) >> 16;
        u1 = (u1 + 0x7FFFu + ((u1 >> 16) & 1u)) >> 16;
        ((u32*)drow)[i >> 1] = u0 | (u1 << 16);
    }
}

// ---------------------------------------------------------------------------
// K2: scores = A·Bᵀ via v_wmma_f32_16x16x32_bf16, fused row max/argmax.
// Block = 8 waves, 32 rows/wave (2 row tiles). B panel staged through LDS in
// 128-token chunks with a software pipeline (global loads issued for chunk
// c+1 while WMMAs consume chunk c). Score matrix never materialized.
// ---------------------------------------------------------------------------
__global__ __launch_bounds__(256) void tome_scores(const u16* __restrict__ A,
                                                   const u16* __restrict__ Bm,
                                                   float* __restrict__ node_max,
                                                   int*   __restrict__ node_idx) {
    __shared__ __align__(16) u16 sb[CHUNK_ * D_];      // 16 KB
    const int tid  = threadIdx.x;
    const int wave = tid >> 5;
    const int l    = tid & 31;
    const int half = l >> 4;       // which 16-lane half
    const int idx  = l & 15;
    const int batch    = blockIdx.x >> 3;              // grid = B*8 blocks
    const int row_base = (blockIdx.x & 7) * 256 + wave * 32;

    // ----- A fragments for two 16-row tiles (16x32 bf16 layout, ISA 7.12.2)
    const int kb = half * 8;       // lanes 0-15: K 0..7 & 16..23 ; 16-31: +8
    v16bf a00, a01, a10, a11;
    {
        const u16* ar0 = A + ((size_t)batch * TA_ + row_base + idx) * D_;
        const u16* ar1 = ar0 + 16 * D_;
        ((v8bf*)&a00)[0] = *(const v8bf*)(ar0 + kb);
        ((v8bf*)&a00)[1] = *(const v8bf*)(ar0 + kb + 16);
        ((v8bf*)&a01)[0] = *(const v8bf*)(ar0 + kb + 32);
        ((v8bf*)&a01)[1] = *(const v8bf*)(ar0 + kb + 48);
        ((v8bf*)&a10)[0] = *(const v8bf*)(ar1 + kb);
        ((v8bf*)&a10)[1] = *(const v8bf*)(ar1 + kb + 16);
        ((v8bf*)&a11)[0] = *(const v8bf*)(ar1 + kb + 32);
        ((v8bf*)&a11)[1] = *(const v8bf*)(ar1 + kb + 48);
    }

    float rm[16];
    int   ri[16];
#pragma unroll
    for (int v = 0; v < 16; ++v) { rm[v] = -__builtin_inff(); ri[v] = 0; }

    // Cooperative staging: thread tid copies 64 B of the chunk
    // (token tid/2, bf16s [ (tid&1)*32 , +32 ) )
    const u16* gsrc = Bm + (size_t)batch * TB_ * D_
                         + (size_t)(tid >> 1) * D_ + (tid & 1) * 32;
    u16* ldst = sb + (tid >> 1) * D_ + (tid & 1) * 32;

    float4 st[4];
#pragma unroll
    for (int q = 0; q < 4; ++q) st[q] = ((const float4*)gsrc)[q];   // chunk 0

    for (int c = 0; c < TB_ / CHUNK_; ++c) {
#pragma unroll
        for (int q = 0; q < 4; ++q) ((float4*)ldst)[q] = st[q];
        __syncthreads();

        if (c < TB_ / CHUNK_ - 1) {          // kick loads for next chunk now
            gsrc += CHUNK_ * D_;
#pragma unroll
            for (int q = 0; q < 4; ++q) st[q] = ((const float4*)gsrc)[q];
        }

#pragma unroll
        for (int jt = 0; jt < CHUNK_ / 16; ++jt) {
            // B fragment (32x16): lane = column N; halves split K; from LDS
            const u16* brow = sb + (jt * 16 + idx) * D_;
            v16bf b0, b1;
            ((v8bf*)&b0)[0] = *(const v8bf*)(brow + half * 8);
            ((v8bf*)&b0)[1] = *(const v8bf*)(brow + half * 8 + 16);
            ((v8bf*)&b1)[0] = *(const v8bf*)(brow + 32 + half * 8);
            ((v8bf*)&b1)[1] = *(const v8bf*)(brow + 32 + half * 8 + 16);

            v8f c0 = {}, c1 = {};
            c0 = __builtin_amdgcn_wmma_f32_16x16x32_bf16(false, a00, false, b0,
                                                         (short)0, c0, false, false);
            c0 = __builtin_amdgcn_wmma_f32_16x16x32_bf16(false, a01, false, b1,
                                                         (short)0, c0, false, false);
            c1 = __builtin_amdgcn_wmma_f32_16x16x32_bf16(false, a10, false, b0,
                                                         (short)0, c1, false, false);
            c1 = __builtin_amdgcn_wmma_f32_16x16x32_bf16(false, a11, false, b1,
                                                         (short)0, c1, false, false);

            const int col = c * CHUNK_ + jt * 16 + idx;
#pragma unroll
            for (int v = 0; v < 8; ++v) {    // lane holds rows M=v+half*8, col
                float s0 = c0[v];
                if (s0 > rm[v])     { rm[v]     = s0; ri[v]     = col; }
                float s1 = c1[v];
                if (s1 > rm[v + 8]) { rm[v + 8] = s1; ri[v + 8] = col; }
            }
        }
        __syncthreads();                     // protect sb before next store
    }

    // scores[:,0,:] = -inf  (global row 0 = tile0 element 0, half 0)
    if (row_base == 0 && half == 0) { rm[0] = -__builtin_inff(); ri[0] = 0; }

    // max/argmax across the 16 lanes of each half (ties -> lowest column)
#pragma unroll
    for (int m = 8; m >= 1; m >>= 1) {
#pragma unroll
        for (int v = 0; v < 16; ++v) {
            float os = __shfl_xor(rm[v], m, 32);
            int   oi = __shfl_xor(ri[v], m, 32);
            if (os > rm[v] || (os == rm[v] && oi < ri[v])) { rm[v] = os; ri[v] = oi; }
        }
    }
    if (idx == 0) {
        const size_t base0 = (size_t)batch * TA_ + row_base + half * 8;
#pragma unroll
        for (int v = 0; v < 8; ++v) {
            node_max[base0 + v]      = rm[v];
            node_idx[base0 + v]      = ri[v];
            node_max[base0 + 16 + v] = rm[v + 8];
            node_idx[base0 + 16 + v] = ri[v + 8];
        }
    }
}

// ---------------------------------------------------------------------------
// K3: per-batch bitonic sorts in LDS.
//   edge_idx = argsort(-node_max) (ties -> index asc, matching stable argsort)
//   src_idx = edge_idx[:r]; dst_idx = node_idx[src_idx]; unm_idx = sort(rest)
// ---------------------------------------------------------------------------
__global__ __launch_bounds__(1024) void tome_sort(const float* __restrict__ node_max,
                                                  const int*   __restrict__ node_idx,
                                                  int* __restrict__ src_idx,
                                                  int* __restrict__ unm_idx,
                                                  int* __restrict__ dst_idx) {
    __shared__ float kv[TA_];
    __shared__ int   iv[TA_];
    __shared__ int   uv[R_];
    const int b = blockIdx.x;
    const int t = threadIdx.x;

    for (int i = t; i < TA_; i += 1024) { kv[i] = node_max[b * TA_ + i]; iv[i] = i; }
    __syncthreads();

    for (int k = 2; k <= TA_; k <<= 1) {
        for (int j = k >> 1; j > 0; j >>= 1) {
            for (int i = t; i < TA_; i += 1024) {
                int ixj = i ^ j;
                if (ixj > i) {
                    bool up = ((i & k) == 0);
                    float va = kv[i], vb = kv[ixj];
                    int   ia = iv[i], ib = iv[ixj];
                    bool aFirst = (va > vb) || (va == vb && ia < ib);
                    if (up ? !aFirst : aFirst) {
                        kv[i] = vb; iv[i] = ib; kv[ixj] = va; iv[ixj] = ia;
                    }
                }
            }
            __syncthreads();
        }
    }

    src_idx[b * R_ + t] = iv[t];
    dst_idx[b * R_ + t] = node_idx[b * TA_ + iv[t]];
    uv[t] = iv[R_ + t];
    __syncthreads();

    for (int k = 2; k <= R_; k <<= 1) {
        for (int j = k >> 1; j > 0; j >>= 1) {
            int ixj = t ^ j;
            if (ixj > t) {
                bool up = ((t & k) == 0);
                int va = uv[t], vb = uv[ixj];
                if (up ? (va > vb) : (va < vb)) { uv[t] = vb; uv[ixj] = va; }
            }
            __syncthreads();
        }
    }
    unm_idx[b * R_ + t] = uv[t];
}

// ---------------------------------------------------------------------------
// K4a: build output rows (unm gather + dst copy), init counts. float4 streams.
// ---------------------------------------------------------------------------
__global__ __launch_bounds__(256) void tome_gather(const float* __restrict__ x,
                                                   const int* __restrict__ unm_idx,
                                                   float* __restrict__ out,
                                                   float* __restrict__ cnt) {
    const int row = blockIdx.x;            // 0 .. B*3072-1
    const int b = row / TOUT_;
    const int r = row - b * TOUT_;
    const float4* srow;
    if (r < R_) {                          // unmerged sources, token 2*u
        int u = unm_idx[b * R_ + r];
        srow = (const float4*)(x + ((size_t)b * T_ + 2 * u) * C_);
    } else {                               // destinations, token 2*j+1
        int j = r - R_;
        srow = (const float4*)(x + ((size_t)b * T_ + 2 * j + 1) * C_);
        if (threadIdx.x == 0) cnt[b * TB_ + j] = 1.0f;
    }
    ((float4*)(out + (size_t)row * C_))[threadIdx.x] = srow[threadIdx.x];
}

// ---------------------------------------------------------------------------
// K4b: scatter-add merged source rows into destination rows (f32 atomics).
// ---------------------------------------------------------------------------
__global__ __launch_bounds__(256) void tome_scatter(const float* __restrict__ x,
                                                    const int* __restrict__ src_idx,
                                                    const int* __restrict__ dst_idx,
                                                    float* __restrict__ out,
                                                    float* __restrict__ cnt) {
    const int e = blockIdx.x;              // 0 .. B*r-1
    const int b = e >> 10;
    const int s = src_idx[e];
    const int d = dst_idx[e];
    const float* srow = x + ((size_t)b * T_ + 2 * s) * C_;
    float*       drow = out + ((size_t)b * TOUT_ + R_ + d) * C_;
    const int i = threadIdx.x * 4;
    atomicAdd(&drow[i + 0], srow[i + 0]);
    atomicAdd(&drow[i + 1], srow[i + 1]);
    atomicAdd(&drow[i + 2], srow[i + 2]);
    atomicAdd(&drow[i + 3], srow[i + 3]);
    if (threadIdx.x == 0) atomicAdd(&cnt[b * TB_ + d], 1.0f);
}

// ---------------------------------------------------------------------------
// K4c: divide destination rows by merge counts.
// ---------------------------------------------------------------------------
__global__ __launch_bounds__(256) void tome_div(float* __restrict__ out,
                                                const float* __restrict__ cnt) {
    const int row = blockIdx.x;            // 0 .. B*2048-1
    const int b = row >> 11;
    const int j = row & (TB_ - 1);
    const float inv = 1.0f / cnt[row];
    float4* drow = (float4*)(out + ((size_t)b * TOUT_ + R_ + j) * C_);
    float4 v = drow[threadIdx.x];
    v.x *= inv; v.y *= inv; v.z *= inv; v.w *= inv;
    drow[threadIdx.x] = v;
}

// ---------------------------------------------------------------------------
extern "C" void kernel_launch(void* const* d_in, const int* in_sizes, int n_in,
                              void* d_out, int out_size, void* d_ws, size_t ws_size,
                              hipStream_t stream) {
    const float* x      = (const float*)d_in[0];
    const float* metric = (const float*)d_in[1];
    (void)in_sizes; (void)n_in; (void)out_size; (void)ws_size;   // r fixed at 1024

    char* ws = (char*)d_ws;
    u16*   Aws      = (u16*)(ws);                               // 8 MB
    u16*   Bws      = (u16*)(ws + (8u  << 20));                 // 8 MB
    float* node_max = (float*)(ws + (16u << 20));               // 256 KB
    int*   node_idx = (int*)  (ws + (16u << 20) + 0x040000);    // 256 KB
    int*   src_idx  = (int*)  (ws + (16u << 20) + 0x080000);    // 128 KB
    int*   unm_idx  = (int*)  (ws + (16u << 20) + 0x0A0000);    // 128 KB
    int*   dst_idx  = (int*)  (ws + (16u << 20) + 0x0C0000);    // 128 KB
    float* cnt      = (float*)(ws + (16u << 20) + 0x0E0000);    // 256 KB
    float* out = (float*)d_out;

    tome_norm   <<<(B_ * T_) / 256, 256, 0, stream>>>(metric, Aws, Bws);
    tome_scores <<<B_ * 8,         256, 0, stream>>>(Aws, Bws, node_max, node_idx);
    tome_sort   <<<B_,            1024, 0, stream>>>(node_max, node_idx,
                                                     src_idx, unm_idx, dst_idx);
    tome_gather <<<B_ * TOUT_,     256, 0, stream>>>(x, unm_idx, out, cnt);
    tome_scatter<<<B_ * R_,        256, 0, stream>>>(x, src_idx, dst_idx, out, cnt);
    tome_div    <<<B_ * TB_,       256, 0, stream>>>(out, cnt);
}
